// Efficient_Self_Attention_42563125904001
// MI455X (gfx1250) — matrix operand order repaired and checked by reading, not compiled, not run on hardware
//
#include <hip/hip_runtime.h>

typedef __bf16 bf16;
typedef __attribute__((ext_vector_type(16))) __bf16 v16bf;
typedef __attribute__((ext_vector_type(8)))  __bf16 v8bf;
typedef __attribute__((ext_vector_type(8)))  float  v8f;
typedef __attribute__((ext_vector_type(4)))  float  v4f;

#define DIMX   1024
#define CDIM   512
#define NHEAD  8
#define HD     64
#define SEQ    4096
#define BATCH  4
#define CHUNK  1024
#define MROWS  (BATCH * SEQ)          // 16384
#define QK_SCALE 0.08838834764831845f // (1024/8)^-0.5

// native f32 -> bf16 convert (v_cvt_pk_bf16_f32 on gfx1250)
__device__ __forceinline__ bf16 f2bf(float x) { return (bf16)x; }

__device__ __forceinline__ v8f wmma_bf16(v16bf a, v16bf b, v8f c) {
  return __builtin_amdgcn_wmma_f32_16x16x32_bf16(false, a, false, b, (short)0, c,
                                                 false, false);
}

// ---------------- weight transpose+convert: W (K, Nc) f32 -> Wt (Nc, K) bf16
__global__ void k_transpose(const float* __restrict__ W, bf16* __restrict__ Wt,
                            int K, int Nc) {
  int idx = blockIdx.x * blockDim.x + threadIdx.x;
  if (idx < K * Nc) {
    int k = idx / Nc;
    int c = idx - k * Nc;
    Wt[(size_t)c * K + k] = f2bf(W[idx]);
  }
}

// load one bf16 A-fragment (rows rbase..rbase+15, k..k+31) from row-major X
__device__ __forceinline__ v16bf load_a_bf16(const bf16* __restrict__ X,
                                             int ldx, int rbase, int k,
                                             int r15, int half) {
  union { v16bf v16; v8bf v8[2]; } u;
  const bf16* xrow = X + (size_t)(rbase + r15) * ldx + k + 16 * half;
  u.v8[0] = *(const v8bf*)(xrow);
  u.v8[1] = *(const v8bf*)(xrow + 8);
  return u.v16;
}

// ---------------- GEMM1: xr = x @ W_reduce  (f32 A converted on load)
// X (16384, 1024) f32, Wt (512, 1024) bf16, Y (16384, 512) bf16
// one wave = 32x64 output tile
__global__ __launch_bounds__(256) void k_gemm_reduce(const float* __restrict__ X,
                                                     const bf16* __restrict__ Wt,
                                                     bf16* __restrict__ Y) {
  const int lane = threadIdx.x & 31;
  const int wid  = (blockIdx.x * 256 + threadIdx.x) >> 5;
  const int rowTile = wid >> 3;   // 512 tiles of 32 rows
  const int colGrp  = wid & 7;    // 8 groups of 64 cols
  const int r15 = lane & 15, half = lane >> 4;
  const float* xrow0 = X + (size_t)(rowTile * 32 + r15) * DIMX;
  const float* xrow1 = xrow0 + (size_t)16 * DIMX;

  v8f z = {0.f,0.f,0.f,0.f,0.f,0.f,0.f,0.f};
  v8f acc[2][4] = {{z, z, z, z}, {z, z, z, z}};
  for (int k = 0; k < DIMX; k += 32) {
    v16bf af0, af1;
    {
      v4f a0 = *(const v4f*)(xrow0 + k + 16 * half);
      v4f a1 = *(const v4f*)(xrow0 + k + 16 * half + 4);
      v4f a2 = *(const v4f*)(xrow0 + k + 16 * half + 8);
      v4f a3 = *(const v4f*)(xrow0 + k + 16 * half + 12);
      v4f b0 = *(const v4f*)(xrow1 + k + 16 * half);
      v4f b1 = *(const v4f*)(xrow1 + k + 16 * half + 4);
      v4f b2 = *(const v4f*)(xrow1 + k + 16 * half + 8);
      v4f b3 = *(const v4f*)(xrow1 + k + 16 * half + 12);
#pragma unroll
      for (int i = 0; i < 4; ++i) {
        af0[i]      = f2bf(a0[i]); af0[i + 4]  = f2bf(a1[i]);
        af0[i + 8]  = f2bf(a2[i]); af0[i + 12] = f2bf(a3[i]);
        af1[i]      = f2bf(b0[i]); af1[i + 4]  = f2bf(b1[i]);
        af1[i + 8]  = f2bf(b2[i]); af1[i + 12] = f2bf(b3[i]);
      }
    }
#pragma unroll
    for (int t = 0; t < 4; ++t) {
      const int col = colGrp * 64 + t * 16 + r15;
      v16bf bfrag = *(const v16bf*)(Wt + (size_t)col * DIMX + k + 16 * half);
      acc[0][t] = wmma_bf16(af0, bfrag, acc[0][t]);
      acc[1][t] = wmma_bf16(af1, bfrag, acc[1][t]);
    }
  }
#pragma unroll
  for (int rt = 0; rt < 2; ++rt)
#pragma unroll
    for (int t = 0; t < 4; ++t)
#pragma unroll
      for (int j = 0; j < 8; ++j) {
        const int rr = rowTile * 32 + rt * 16 + j + 8 * half;
        const int cc = colGrp * 64 + t * 16 + r15;
        Y[(size_t)rr * CDIM + cc] = f2bf(acc[rt][t][j]);
      }
}

// ---------------- GEMM2: qkv = xr @ W_qkv, scatter to q(scaled)/k/(v transposed)
// X (16384,512) bf16, Wt (1536,512) bf16; one wave = 32x64 tile
__global__ __launch_bounds__(256) void k_gemm_qkv(const bf16* __restrict__ X,
                                                  const bf16* __restrict__ Wt,
                                                  bf16* __restrict__ qb,
                                                  bf16* __restrict__ kb,
                                                  bf16* __restrict__ vb) {
  const int lane = threadIdx.x & 31;
  const int wid  = (blockIdx.x * 256 + threadIdx.x) >> 5;
  const int rowTile = wid / 24;   // 512 tiles of 32 rows
  const int colGrp  = wid % 24;   // 24 groups of 64 (1536 cols)
  const int r15 = lane & 15, half = lane >> 4;

  v8f z = {0.f,0.f,0.f,0.f,0.f,0.f,0.f,0.f};
  v8f acc[2][4] = {{z, z, z, z}, {z, z, z, z}};
  for (int k = 0; k < CDIM; k += 32) {
    v16bf af0 = load_a_bf16(X, CDIM, rowTile * 32,      k, r15, half);
    v16bf af1 = load_a_bf16(X, CDIM, rowTile * 32 + 16, k, r15, half);
#pragma unroll
    for (int t = 0; t < 4; ++t) {
      const int col = colGrp * 64 + t * 16 + r15;
      v16bf bfrag = *(const v16bf*)(Wt + (size_t)col * CDIM + k + 16 * half);
      acc[0][t] = wmma_bf16(af0, bfrag, acc[0][t]);
      acc[1][t] = wmma_bf16(af1, bfrag, acc[1][t]);
    }
  }
#pragma unroll
  for (int rt = 0; rt < 2; ++rt)
#pragma unroll
    for (int t = 0; t < 4; ++t)
#pragma unroll
      for (int j = 0; j < 8; ++j) {
        const int rr = rowTile * 32 + rt * 16 + j + 8 * half;   // global row (B*N)
        const int cc = colGrp * 64 + t * 16 + r15;              // 0..1535
        const int which = cc / CDIM;
        const int c2 = cc - which * CDIM;
        const int h  = c2 / HD;
        const int d  = c2 - h * HD;
        const int b  = rr >> 12;
        const int ni = rr & (SEQ - 1);
        const float val = acc[rt][t][j];
        if (which == 0)
          qb[((size_t)((b * NHEAD + h) * SEQ + ni)) * HD + d] = f2bf(val * QK_SCALE);
        else if (which == 1)
          kb[((size_t)((b * NHEAD + h) * SEQ + ni)) * HD + d] = f2bf(val);
        else
          vb[((size_t)((b * NHEAD + h) * HD + d)) * SEQ + ni] = f2bf(val);
      }
}

// ---------------- chunked flash attention: one (b,h,seg, 64-row q tile) per block
// q,k: (B,H,N,hd) bf16 (q pre-scaled); v: (B,H,hd,N) bf16; o: (B*N, 512) bf16
__global__ __launch_bounds__(128) void k_attn(const bf16* __restrict__ qb,
                                              const bf16* __restrict__ kbuf,
                                              const bf16* __restrict__ vb,
                                              bf16* __restrict__ ob) {
  __shared__ __align__(32) float s_lds[4][16][32];
  __shared__ __align__(32) bf16  p_lds[4][16][32];
  __shared__ float f_lds[4][16];

  const int lane = threadIdx.x & 31;
  const int w    = threadIdx.x >> 5;
  const int r15  = lane & 15, half = lane >> 4;
  const int qt    = blockIdx.x & 15;      // 16 q tiles per chunk
  const int chunk = blockIdx.x >> 4;      // (b,h,s)
  const int s = chunk & 3, h = (chunk >> 2) & 7, b = chunk >> 5;
  const int bh = b * NHEAD + h;
  const int qrow0 = s * CHUNK + qt * 64 + w * 16;  // global n-index of wave tile

  // Q A-fragments (d = 0..31 and 32..63)
  const bf16* qrow = qb + ((size_t)(bh * SEQ + qrow0 + r15)) * HD;
  const int dh = 16 * half;
  union { v16bf v16; v8bf v8[2]; } u;
  u.v8[0] = *(const v8bf*)(qrow + dh);
  u.v8[1] = *(const v8bf*)(qrow + 8 + dh);
  v16bf qa0 = u.v16;
  u.v8[0] = *(const v8bf*)(qrow + 32 + dh);
  u.v8[1] = *(const v8bf*)(qrow + 40 + dh);
  v16bf qa1 = u.v16;

  v8f z = {0.f,0.f,0.f,0.f,0.f,0.f,0.f,0.f};
  v8f o0 = z, o1 = z, o2 = z, o3 = z;
  float m = -1e30f, l = 0.f;

  for (int kt = 0; kt < 32; ++kt) {
    const int kbase = s * CHUNK + kt * 32;
    // ---- S = Q @ K^T (16x32 scores, scale pre-folded into q)
    const bf16* kp0 = kbuf + ((size_t)(bh * SEQ + kbase + r15)) * HD + 16 * half;
    const bf16* kp1 = kbuf + ((size_t)(bh * SEQ + kbase + 16 + r15)) * HD + 16 * half;
    v8f s0 = wmma_bf16(qa0, *(const v16bf*)kp0, z);
    s0     = wmma_bf16(qa1, *(const v16bf*)(kp0 + 32), s0);
    v8f s1 = wmma_bf16(qa0, *(const v16bf*)kp1, z);
    s1     = wmma_bf16(qa1, *(const v16bf*)(kp1 + 32), s1);
#pragma unroll
    for (int j = 0; j < 8; ++j) {           // C layout -> row-major LDS
      s_lds[w][j + 8 * half][r15]      = s0[j];
      s_lds[w][j + 8 * half][16 + r15] = s1[j];
    }
    __syncthreads();
    // ---- online softmax stats, one row per lane (lanes 0..15)
    if (lane < 16) {
      float tmax = -1e30f;
#pragma unroll
      for (int e = 0; e < 32; ++e) tmax = fmaxf(tmax, s_lds[w][r15][e]);
      const float mn   = fmaxf(m, tmax);
      const float corr = __expf(m - mn);
      float psum = 0.f;
#pragma unroll
      for (int e = 0; e < 32; ++e) {
        const float p = __expf(s_lds[w][r15][e] - mn);
        psum += p;
        p_lds[w][r15][e] = f2bf(p);
      }
      f_lds[w][r15] = corr;
      l = l * corr + psum;
      m = mn;
    }
    __syncthreads();
    // ---- rescale O accumulators by per-row correction
#pragma unroll
    for (int j = 0; j < 8; ++j) {
      const float f = f_lds[w][j + 8 * half];
      o0[j] *= f; o1[j] *= f; o2[j] *= f; o3[j] *= f;
    }
    // ---- P A-fragment from LDS
    u.v8[0] = *(const v8bf*)(&p_lds[w][r15][dh]);
    u.v8[1] = *(const v8bf*)(&p_lds[w][r15][8 + dh]);
    v16bf pa = u.v16;
    // ---- O += P @ V (V transposed -> contiguous B-fragment loads)
    const bf16* vp = vb + ((size_t)(bh * HD + r15)) * SEQ + kbase + 16 * half;
    o0 = wmma_bf16(pa, *(const v16bf*)(vp),                    o0);
    o1 = wmma_bf16(pa, *(const v16bf*)(vp + 16 * (size_t)SEQ), o1);
    o2 = wmma_bf16(pa, *(const v16bf*)(vp + 32 * (size_t)SEQ), o2);
    o3 = wmma_bf16(pa, *(const v16bf*)(vp + 48 * (size_t)SEQ), o3);
    __syncthreads();
  }

  if (lane < 16) f_lds[w][r15] = 1.0f / l;
  __syncthreads();
#pragma unroll
  for (int j = 0; j < 8; ++j) {
    const float inv = f_lds[w][j + 8 * half];
    const size_t rr = (size_t)(b * SEQ + qrow0 + j + 8 * half);
    bf16* orow = ob + rr * CDIM + h * HD;
    orow[ 0 + r15] = f2bf(o0[j] * inv);
    orow[16 + r15] = f2bf(o1[j] * inv);
    orow[32 + r15] = f2bf(o2[j] * inv);
    orow[48 + r15] = f2bf(o3[j] * inv);
  }
}

// ---------------- GEMM3: out = O @ W_proj + b  (f32 out); one wave = 32x64 tile
// X (16384,512) bf16, Wt (1024,512) bf16, out (16384,1024) f32
__global__ __launch_bounds__(256) void k_gemm_proj(const bf16* __restrict__ X,
                                                   const bf16* __restrict__ Wt,
                                                   const float* __restrict__ bias,
                                                   float* __restrict__ out) {
  const int lane = threadIdx.x & 31;
  const int wid  = (blockIdx.x * 256 + threadIdx.x) >> 5;
  const int rowTile = wid >> 4;   // 512 tiles of 32 rows
  const int colGrp  = wid & 15;   // 16 groups of 64 (1024 cols)
  const int r15 = lane & 15, half = lane >> 4;

  v8f z = {0.f,0.f,0.f,0.f,0.f,0.f,0.f,0.f};
  v8f acc[2][4] = {{z, z, z, z}, {z, z, z, z}};
  for (int k = 0; k < CDIM; k += 32) {
    v16bf af0 = load_a_bf16(X, CDIM, rowTile * 32,      k, r15, half);
    v16bf af1 = load_a_bf16(X, CDIM, rowTile * 32 + 16, k, r15, half);
#pragma unroll
    for (int t = 0; t < 4; ++t) {
      const int col = colGrp * 64 + t * 16 + r15;
      v16bf bfrag = *(const v16bf*)(Wt + (size_t)col * CDIM + k + 16 * half);
      acc[0][t] = wmma_bf16(af0, bfrag, acc[0][t]);
      acc[1][t] = wmma_bf16(af1, bfrag, acc[1][t]);
    }
  }
#pragma unroll
  for (int rt = 0; rt < 2; ++rt)
#pragma unroll
    for (int t = 0; t < 4; ++t)
#pragma unroll
      for (int j = 0; j < 8; ++j) {
        const int rr = rowTile * 32 + rt * 16 + j + 8 * half;
        const int cc = colGrp * 64 + t * 16 + r15;
        out[(size_t)rr * DIMX + cc] = acc[rt][t][j] + bias[cc];
      }
}

extern "C" void kernel_launch(void* const* d_in, const int* in_sizes, int n_in,
                              void* d_out, int out_size, void* d_ws, size_t ws_size,
                              hipStream_t stream) {
  (void)in_sizes; (void)n_in; (void)out_size; (void)ws_size;
  const float* x       = (const float*)d_in[0];
  const float* Wreduce = (const float*)d_in[1];   // (1024, 512)
  const float* Wqkv    = (const float*)d_in[2];   // (512, 1536)
  const float* Wproj   = (const float*)d_in[3];   // (512, 1024)
  const float* bproj   = (const float*)d_in[4];   // (1024,)
  float* out = (float*)d_out;

  char* ws = (char*)d_ws;
  size_t off = 0;
  auto carve = [&](size_t bytes) -> void* {
    void* p = ws + off;
    off += (bytes + 255) & ~(size_t)255;
    return p;
  };
  bf16* wr_t = (bf16*)carve((size_t)CDIM * DIMX * 2);       // (512,1024)
  bf16* wq_t = (bf16*)carve((size_t)3 * CDIM * CDIM * 2);   // (1536,512)
  bf16* wp_t = (bf16*)carve((size_t)DIMX * CDIM * 2);       // (1024,512)
  bf16* xr   = (bf16*)carve((size_t)MROWS * CDIM * 2);
  bf16* qb   = (bf16*)carve((size_t)MROWS * CDIM * 2);
  bf16* kb   = (bf16*)carve((size_t)MROWS * CDIM * 2);
  bf16* vb   = (bf16*)carve((size_t)MROWS * CDIM * 2);
  bf16* ob   = (bf16*)carve((size_t)MROWS * CDIM * 2);

  int tot;
  tot = DIMX * CDIM;
  k_transpose<<<(tot + 255) / 256, 256, 0, stream>>>(Wreduce, wr_t, DIMX, CDIM);
  tot = CDIM * 3 * CDIM;
  k_transpose<<<(tot + 255) / 256, 256, 0, stream>>>(Wqkv, wq_t, CDIM, 3 * CDIM);
  tot = CDIM * DIMX;
  k_transpose<<<(tot + 255) / 256, 256, 0, stream>>>(Wproj, wp_t, CDIM, DIMX);

  k_gemm_reduce<<<512, 256, 0, stream>>>(x, wr_t, xr);          // 4096 waves, 32x64 tiles
  k_gemm_qkv  <<<1536, 256, 0, stream>>>(xr, wq_t, qb, kb, vb); // 12288 waves
  k_attn      <<<2048, 128, 0, stream>>>(qb, kb, vb, ob);       // (b,h,s) x 16 tiles
  k_gemm_proj <<<1024, 256, 0, stream>>>(ob, wp_t, bproj, out); // 8192 waves
}
